// KPConv_68281390072586
// MI455X (gfx1250) — matrix operand order repaired and checked by reading, not compiled, not run on hardware
//
#include <hip/hip_runtime.h>
#include <hip/hip_bf16.h>

typedef __attribute__((ext_vector_type(16))) __bf16 v16bf;
typedef __attribute__((ext_vector_type(8)))  float  v8f;

#define BB   2
#define NN   16384
#define MM   16384
#define KK   32
#define CIN  64
#define COUT 128
#define KS   15
// sigma = 0.5 * 0.1 = 0.05 ; 1/(2*sigma^2) = 200 ; folded with log2(e):
#define NEG_SCALE_LOG2E (-288.53900817779268f)   // -200 / ln(2)

// Fast 2^x -> single v_exp_f32 where the builtin exists.
__device__ __forceinline__ float fast_exp2(float x) {
#if __has_builtin(__builtin_amdgcn_exp2f)
  return __builtin_amdgcn_exp2f(x);
#else
  return exp2f(x);
#endif
}

// RNE f32->bf16 (used only in the one-shot weight pre-pass).
__device__ __forceinline__ unsigned short f2bf_rne(float x) {
  unsigned u = __float_as_uint(x);
  u += 0x7FFFu + ((u >> 16) & 1u);
  return (unsigned short)(u >> 16);
}

// Truncating f32->bf16: upper half of the f32 bits. Zero VALU when paired
// with a 16-bit store (compiler selects ds_store_b16_d16_hi).
__device__ __forceinline__ unsigned short f2bf_trunc(float x) {
  return (unsigned short)(__float_as_uint(x) >> 16);
}

// Pack two truncated bf16 into one dword: {bf16(hi), bf16(lo)}.
__device__ __forceinline__ unsigned pack_bf16x2(float lo, float hi) {
  return (__float_as_uint(hi) & 0xFFFF0000u) | (__float_as_uint(lo) >> 16);
}

union BfA { int i[8]; v16bf v; };

// A-matrix (16 x 32 bf16) operand per CDNA5 ISA layout, loaded from a
// row-major [16][rowStride] array. colOff selects the K-chunk (multiple of 32).
// lane 0-15: M=lane, K-half {0..7,16..23}; lane 16-31: M=lane-16, +8.
__device__ __forceinline__ v16bf load_A(const unsigned short* base, int rowStride,
                                        int lane, int colOff) {
  BfA u;
  int m = lane & 15;
  int khalf = (lane < 16) ? 0 : 8;
  const int* rp = (const int*)(base + (size_t)m * rowStride);
#pragma unroll
  for (int v = 0; v < 8; ++v) {
    int k = ((v < 4) ? (2 * v) : (16 + 2 * (v - 4))) + khalf + colOff;
    u.i[v] = rp[k >> 1];
  }
  return u.v;
}

// B-matrix (32 x 16 bf16) operand: element (k, n) stored transposed at
// base[n*nStride + kOff + k] (contiguous in k -> dword = K pair).
// lane 0-15: N=lane, K {0..7,16..23} (2/VGPR); lane 16-31: N=lane-16, +8.
__device__ __forceinline__ v16bf load_B(const unsigned short* base, int nStride,
                                        int lane, int nOff, int kOff) {
  BfA u;
  int n = (lane & 15) + nOff;
  int kh2 = (lane < 16) ? 0 : 4;   // dword offset of the K-half (+8 k)
  const int* rp = (const int*)(base + (size_t)n * nStride + kOff);
#pragma unroll
  for (int v = 0; v < 8; ++v) u.i[v] = rp[((v < 4) ? v : (v + 4)) + kh2];
  return u.v;
}

// Pre-pass: W[j][c][o] (f32) -> Wt[o][j*64+c] (bf16 RNE), j=15 row zeroed.
__global__ void prep_weights(const float* __restrict__ W,
                             unsigned short* __restrict__ Wt) {
  int id = blockIdx.x * 256 + threadIdx.x;   // COUT*1024 = 131072 total
  int o  = id >> 10;
  int jc = id & 1023;
  int j = jc >> 6, c = jc & 63;
  float v = (j < KS) ? W[((size_t)j * CIN + c) * COUT + o] : 0.f;
  Wt[(size_t)o * 1024 + jc] = f2bf_rne(v);
}

__global__ void __launch_bounds__(256)
kpconv_kernel(const float* __restrict__ query_points,
              const float* __restrict__ support_points,
              const float* __restrict__ support_features,
              const int*   __restrict__ neighbor_idx,
              const float* __restrict__ kernel_points,
              const unsigned short* __restrict__ Wt,  // [COUT][1024] bf16
              const float* __restrict__ bias,
              float* __restrict__ out) {
  __shared__ unsigned short sh_featsT[16][CIN][KK];  // [q][c][k]  64 KB
  __shared__ unsigned short sh_infl[16][16][KK];     // [q][j][k]  16 KB
  __shared__ unsigned short sh_G[16][1024];          // [q][j*64+c] 32 KB
  __shared__ int   sh_idx[512];
  __shared__ float sh_kp[KS][3];

  const int t    = threadIdx.x;
  const int lane = t & 31;
  const int wave = t >> 5;
  const int base = blockIdx.x * 16;     // flattened b*N + n tile base
  const int b    = base >> 14;          // N == 16384
  const int n0   = base & (NN - 1);

  if (t < KS * 3) ((float*)sh_kp)[t] = kernel_points[t];

  // Warm this wave's weight panel into cache while phase 1 runs.
  {
    int o = wave * 16 + (lane & 15);
    const unsigned short* wrow = Wt + (size_t)o * 1024;
#pragma unroll
    for (int pb = 0; pb < 8; ++pb) __builtin_prefetch(wrow + pb * 128, 0, 0);
  }
  __syncthreads();

  // ---- Phase 1a: influences (exp + normalize) -> LDS bf16 A panels ----
#pragma unroll
  for (int pi = 0; pi < 2; ++pi) {
    int p = t * 2 + pi;                 // 512 (q,k) pairs
    int q = p >> 5, k = p & 31;
    int n = n0 + q;
    int idx = neighbor_idx[((size_t)b * NN + n) * KK + k];
    sh_idx[p] = idx;
    const float* qp = query_points + ((size_t)b * NN + n) * 3;
    const float* sp = support_points + ((size_t)b * MM + idx) * 3;
    float rx = sp[0] - qp[0], ry = sp[1] - qp[1], rz = sp[2] - qp[2];
    float w[KS], sum = 0.f;
#pragma unroll
    for (int j = 0; j < KS; ++j) {
      float dx = rx - sh_kp[j][0];
      float dy = ry - sh_kp[j][1];
      float dz = rz - sh_kp[j][2];
      float d2 = dx * dx + dy * dy + dz * dz;
      float e = fast_exp2(d2 * NEG_SCALE_LOG2E);   // exp(-200*d2)
      w[j] = e;
      sum += e;
    }
    float inv = 1.f / (sum + 1e-6f);
#pragma unroll
    for (int j = 0; j < KS; ++j) sh_infl[q][j][k] = f2bf_trunc(w[j] * inv);
    sh_infl[q][15][k] = 0;              // pad row -> zero G row 15
  }
  __syncthreads();

  // ---- Phase 1b: gather neighbor features -> LDS transposed bf16 ------
  // Each 16-lane group owns a pair of neighbor rows (k0, k0+1): the two
  // bf16 values sharing an LDS dword live in one thread -> ds_store_b32.
  // Global loads stay coalesced: 16 consecutive lanes sweep one 256 B row.
  {
    int lane16 = t & 15;
    for (int pp = (t >> 4); pp < 256; pp += 16) {  // 256 row pairs
      int q  = pp >> 4;
      int k0 = (pp & 15) * 2;
      int idx0 = sh_idx[q * KK + k0];
      int idx1 = sh_idx[q * KK + k0 + 1];
      const float4 f0 = *(const float4*)(support_features +
                          ((size_t)b * MM + idx0) * CIN + lane16 * 4);
      const float4 f1 = *(const float4*)(support_features +
                          ((size_t)b * MM + idx1) * CIN + lane16 * 4);
      int c = lane16 * 4;
      int* rowp = (int*)&sh_featsT[q][c][0];       // stride KK shorts per c
      rowp[(0 * KK + k0) >> 1] = pack_bf16x2(f0.x, f1.x);
      rowp[(1 * KK + k0) >> 1] = pack_bf16x2(f0.y, f1.y);
      rowp[(2 * KK + k0) >> 1] = pack_bf16x2(f0.z, f1.z);
      rowp[(3 * KK + k0) >> 1] = pack_bf16x2(f0.w, f1.w);
    }
  }
  __syncthreads();

  // ---- Phase 2: G[q] = infl(16x32) @ feats(32x64), WMMA per c-tile ----
#pragma unroll
  for (int qi = 0; qi < 2; ++qi) {
    int q = wave * 2 + qi;
    v16bf a = load_A(&sh_infl[q][0][0], KK, lane, 0);
#pragma unroll
    for (int ct = 0; ct < 4; ++ct) {
      v16bf bm = load_B(&sh_featsT[q][0][0], KK, lane, ct * 16, 0);
      v8f d = {};
      d = __builtin_amdgcn_wmma_f32_16x16x32_bf16(false, a, false, bm,
                                                  (short)0, d, false, false);
      int c  = (lane & 15) + ct * 16;   // D layout: N = lane%16
      int jb = (lane >> 4) * 8;         // D layout: M = r + 8*(lane/16)
#pragma unroll
      for (int r = 0; r < 8; ++r)
        sh_G[q][(jb + r) * CIN + c] = f2bf_trunc(d[r]);  // ds_store_b16_d16_hi
    }
  }
  __syncthreads();

  // ---- Phase 3: out[16q x 16o] = G(16x1024) @ W(1024x16), per-wave ----
  {
    int o = wave * 16 + (lane & 15);
    v8f acc = {};
#pragma unroll 4
    for (int kt = 0; kt < 32; ++kt) {
      v16bf a  = load_A(&sh_G[0][0], 1024, lane, kt * 32);
      v16bf bw = load_B(Wt, 1024, lane, wave * 16, kt * 32);
      acc = __builtin_amdgcn_wmma_f32_16x16x32_bf16(false, a, false, bw,
                                                    (short)0, acc, false, false);
    }
    float bias_o = bias[o];
#pragma unroll
    for (int r = 0; r < 8; ++r) {
      int q = r + (lane >> 4) * 8;
      out[(size_t)(base + q) * COUT + o] = acc[r] + bias_o;
    }
  }
}

extern "C" void kernel_launch(void* const* d_in, const int* in_sizes, int n_in,
                              void* d_out, int out_size, void* d_ws, size_t ws_size,
                              hipStream_t stream) {
  const float* query_points     = (const float*)d_in[0];
  const float* support_points   = (const float*)d_in[1];
  const float* support_features = (const float*)d_in[2];
  const int*   neighbor_idx     = (const int*)d_in[3];
  const float* kernel_points    = (const float*)d_in[4];
  const float* weights          = (const float*)d_in[5];
  const float* bias             = (const float*)d_in[6];
  unsigned short* Wt = (unsigned short*)d_ws;   // 128*1024 bf16 = 256 KB

  prep_weights<<<(COUT * 1024) / 256, 256, 0, stream>>>(weights, Wt);
  kpconv_kernel<<<(BB * NN) / 16, 256, 0, stream>>>(
      query_points, support_points, support_features, neighbor_idx,
      kernel_points, Wt, bias, (float*)d_out);
}
